// CeNNFrontEnd2DG_78262894067925
// MI455X (gfx1250) — matrix operand order repaired and checked by reading, not compiled, not run on hardware
//
#include <hip/hip_runtime.h>
#include <hip/hip_bf16.h>

typedef __attribute__((ext_vector_type(16))) _Float16 v16h;
typedef __attribute__((ext_vector_type(8)))  _Float16 v8h;
typedef __attribute__((ext_vector_type(8)))  float    v8f;
typedef __attribute__((ext_vector_type(4)))  float    v4f;
typedef __attribute__((ext_vector_type(4)))  int      v4i;
typedef __attribute__((address_space(1))) v4i* gp_v4i;   // global int4*
typedef __attribute__((address_space(3))) v4i* lp_v4i;   // LDS int4*

#define HH 512
#define WW 512
#define BB 4
#define CC 64
#define TILE_W 128
#define TILE_H 4
#define NPIX ((size_t)BB * HH * WW)

// Fast tanh: native v_tanh_f32 if the builtin exists, else (1-e)/(1+e) with
// native exp (6 VALU ops instead of ~20 for ocml tanhf).
__device__ __forceinline__ float act_tanh(float x) {
#if __has_builtin(__builtin_amdgcn_tanhf)
    return __builtin_amdgcn_tanhf(x);
#else
    float e = __expf(-2.0f * __builtin_fabsf(x));
    float r = (1.0f - e) / (1.0f + e);
    return __builtin_copysignf(r, x);
#endif
}

// ---------------------------------------------------------------------------
// Pack W_A (with diagonal center-tap clamp) into per-lane WMMA A-fragments.
// Fragment index f = (tap*2 + kb)*4 + mb. Contiguous k order (same as the B loads):
// lane<16 holds row M=lane&15, K = {0..15}; lane>=16 holds K={16..31}.
// ---------------------------------------------------------------------------
__global__ void prep_weights(const float* __restrict__ WA, _Float16* __restrict__ wpk) {
    int gid = blockIdx.x * blockDim.x + threadIdx.x;
    if (gid >= 72 * 32) return;
    int lane = gid & 31, frag = gid >> 5;
    int mb = frag & 3, kb = (frag >> 2) & 1, tap = frag >> 3;
    int m = (lane & 15) + mb * 16;          // output channel
    int base = (lane < 16) ? 0 : 16;
    v16h v;
#pragma unroll
    for (int e = 0; e < 16; ++e) {
        int K = base + e;
        int cin = kb * 32 + K;
        float w = WA[((size_t)m * 64 + cin) * 9 + tap];
        if (m == cin && tap == 4) w = fmaxf(w, 1.0f);   // clamp center self-tap
        v[e] = (_Float16)w;
    }
    *(v16h*)(wpk + (size_t)gid * 16) = v;
}

// ---------------------------------------------------------------------------
// control = conv3x3(u, W_B) (CIN=1), stored NHWC f32. Also x0 = control and
// t0 = tanh(control) as f16 NHWC (first recurrent input).
// ---------------------------------------------------------------------------
__global__ void control_init(const float* __restrict__ u, const float* __restrict__ WB,
                             float* __restrict__ ctl, float* __restrict__ xbuf,
                             _Float16* __restrict__ t0) {
    size_t gid = (size_t)blockIdx.x * blockDim.x + threadIdx.x;
    int q = (int)(gid & 3);
    size_t p = gid >> 2;                    // pixel index
    int x = (int)(p & (WW - 1));
    int y = (int)((p >> 9) & (HH - 1));
    int b = (int)(p >> 18);
    float uv[9];
#pragma unroll
    for (int ky = 0; ky < 3; ++ky)
#pragma unroll
        for (int kx = 0; kx < 3; ++kx) {
            int gy = y + ky - 1, gx = x + kx - 1;
            bool in = (gy >= 0) && (gy < HH) && (gx >= 0) && (gx < WW);
            uv[ky * 3 + kx] = in ? u[((size_t)b * HH + gy) * WW + gx] : 0.0f;
        }
    int c0 = q * 16;
#pragma unroll 4
    for (int cc = 0; cc < 16; ++cc) {
        int c = c0 + cc;
        float s = 0.0f;
#pragma unroll
        for (int k = 0; k < 9; ++k) s += WB[c * 9 + k] * uv[k];
        ctl[p * 64 + c]  = s;
        xbuf[p * 64 + c] = s;
        t0[p * 64 + c]   = (_Float16)act_tanh(s);
    }
}

// ---------------------------------------------------------------------------
// One recurrent step: feedback = conv3x3(t_in, Wpk) via WMMA implicit GEMM,
// x = alpha*x + beta*(feedback + ctl + bias) in place, t_out = tanh(x) f16.
// Block: 256 threads (8 waves), tile = 4 rows x 128 cols. Each wave owns one
// row half (64 px): two pair-iterations of two 16-pixel N-tiles. A fragments
// are reloaded per pair (opaque-offset defeats LICM -> no scratch spills) and
// each A fragment feeds two WMMAs (both N-tiles).
// ---------------------------------------------------------------------------
#define WMMA(A, B, C) __builtin_amdgcn_wmma_f32_16x16x32_f16(false, (A), false, (B), (short)0, (C), false, false)

#define EPILOGUE(ACC, MB)                                                          \
    {                                                                              \
        const int c0 = (MB) * 16 + hi * 8;                                         \
        v4f xa = *(const v4f*)(xq0 + c0);                                          \
        v4f xb = *(const v4f*)(xq0 + c0 + 4);                                      \
        v4f ca = *(const v4f*)(cq0 + c0);                                          \
        v4f cb = *(const v4f*)(cq0 + c0 + 4);                                      \
        v4f ba = *(const v4f*)(bias + c0);                                         \
        v4f bb = *(const v4f*)(bias + c0 + 4);                                     \
        v4f r0, r1; v8h th;                                                        \
        _Pragma("unroll")                                                          \
        for (int j = 0; j < 4; ++j) {                                              \
            float xn = alpha * xa[j] + beta * ((ACC)[j] + ca[j] + ba[j]);          \
            r0[j] = xn; th[j] = (_Float16)act_tanh(xn);                            \
        }                                                                          \
        _Pragma("unroll")                                                          \
        for (int j = 0; j < 4; ++j) {                                              \
            float xn = alpha * xb[j] + beta * ((ACC)[j + 4] + cb[j] + bb[j]);      \
            r1[j] = xn; th[j + 4] = (_Float16)act_tanh(xn);                        \
        }                                                                          \
        *(v4f*)(xq0 + c0)     = r0;                                                \
        *(v4f*)(xq0 + c0 + 4) = r1;                                                \
        *(v8h*)(tout + pix * 64 + c0) = th;                                        \
    }

__global__ void __launch_bounds__(256) cenn_step(
    const _Float16* __restrict__ tin, _Float16* __restrict__ tout,
    float* __restrict__ xbuf, const float* __restrict__ ctl,
    const float* __restrict__ bias, const float* __restrict__ alpha_logit,
    const _Float16* __restrict__ wpk) {
    extern __shared__ _Float16 smem[];  // 6 * 130 * 64 halfs = 99840 B

    const int x0 = blockIdx.x * TILE_W;
    const int y0 = blockIdx.y * TILE_H;
    const int b  = blockIdx.z;

    // Stage tanh(x) halo tile into LDS (zero-padded SAME border), NHWC f16.
#if __has_builtin(__builtin_amdgcn_global_load_async_to_lds_b128)
    // CDNA5 async DMA path: GLOBAL_LOAD_ASYNC_TO_LDS_B128 writes LDS directly
    // (ASYNCcnt-tracked), no VGPR round trip. 16B per lane per transfer.
    for (int idx = threadIdx.x; idx < 6 * 130 * 8; idx += 256) {
        int slot = idx >> 3, i = idx & 7;
        int r = slot / 130, cpx = slot - r * 130;
        int gy = y0 + r - 1, gx = x0 + cpx - 1;
        _Float16* ldst = smem + (size_t)slot * 64 + i * 8;
        if (gy >= 0 && gy < HH && gx >= 0 && gx < WW) {
            const _Float16* gsrc = tin + (((size_t)b * HH + gy) * WW + gx) * 64 + i * 8;
            __builtin_amdgcn_global_load_async_to_lds_b128(
                (gp_v4i)(__attribute__((address_space(1))) void*)(void*)const_cast<_Float16*>(gsrc),
                (lp_v4i)(__attribute__((address_space(3))) void*)(void*)ldst, 0, 0);
        } else {
            *(uint4*)ldst = make_uint4(0u, 0u, 0u, 0u);
        }
    }
#if __has_builtin(__builtin_amdgcn_s_wait_asynccnt)
    __builtin_amdgcn_s_wait_asynccnt(0);
#else
    asm volatile("s_wait_asynccnt 0" ::: "memory");
#endif
#else
    for (int s = threadIdx.x; s < 6 * 130; s += 256) {
        int r = s / 130, cpx = s - r * 130;
        int gy = y0 + r - 1, gx = x0 + cpx - 1;
        uint4* dst = (uint4*)(smem + (size_t)s * 64);
        if (gy >= 0 && gy < HH && gx >= 0 && gx < WW) {
            const uint4* src = (const uint4*)(tin + (((size_t)b * HH + gy) * WW + gx) * 64);
#pragma unroll
            for (int i = 0; i < 8; ++i) dst[i] = src[i];
        } else {
            uint4 z = make_uint4(0u, 0u, 0u, 0u);
#pragma unroll
            for (int i = 0; i < 8; ++i) dst[i] = z;
        }
    }
#endif
    __syncthreads();

    const int lane  = threadIdx.x & 31;
    const int wave  = threadIdx.x >> 5;
    const int wrow  = wave >> 1;          // 0..3 : output row within tile
    const int wcolb = (wave & 1) * 64;    // 0 or 64 : column half
    const int n  = lane & 15;             // pixel within 16-wide N tile
    const int hi = lane >> 4;             // channel half selector

    const float alpha = 1.0f / (1.0f + __expf(-alpha_logit[0]));
    const float beta  = 1.0f - alpha;

    const int y = y0 + wrow;
    for (int ntp = 0; ntp < 2; ++ntp) {
        const int nt0 = ntp * 2;
        const int px0 = x0 + wcolb + nt0 * 16 + n;
        const size_t pixA = ((size_t)b * HH + y) * WW + px0;
        const size_t pixB = pixA + 16;
        __builtin_prefetch(ctl  + pixA * 64, 0, 3);
        __builtin_prefetch(xbuf + pixA * 64, 0, 3);
        __builtin_prefetch(ctl  + pixB * 64, 0, 3);
        __builtin_prefetch(xbuf + pixB * 64, 0, 3);

        // Opaque zero offset: A-fragment loads depend on a volatile asm output
        // so they cannot be hoisted out of the ntp loop (no spills), while the
        // base pointer keeps its inferred global address space (saddr form).
        size_t wofs = 0;
        asm volatile("" : "+s"(wofs));
        const v16h* wb = (const v16h*)(wpk + wofs);

        v8f p0 = {}, p1 = {}, p2 = {}, p3 = {};   // N-tile 0 accumulators
        v8f q0 = {}, q1 = {}, q2 = {}, q3 = {};   // N-tile 1 accumulators
        const int colb = wcolb + nt0 * 16 + n;
#pragma unroll
        for (int tap = 0; tap < 9; ++tap) {
            const int ky = tap / 3, kx = tap - ky * 3;
            const int r = wrow + ky;
#pragma unroll
            for (int kb = 0; kb < 2; ++kb) {
                const size_t boff = (size_t)(r * 130 + colb + kx) * 64 + kb * 32 + hi * 16;
                const v16h bf0 = *(const v16h*)(smem + boff);
                const v16h bf1 = *(const v16h*)(smem + boff + 16 * 64);
                const v16h* ap = wb + ((size_t)(tap * 2 + kb) * 4) * 32 + lane;
                const v16h a0 = ap[0];
                const v16h a1 = ap[32];
                const v16h a2 = ap[64];
                const v16h a3 = ap[96];
                p0 = WMMA(a0, bf0, p0);  q0 = WMMA(a0, bf1, q0);
                p1 = WMMA(a1, bf0, p1);  q1 = WMMA(a1, bf1, q1);
                p2 = WMMA(a2, bf0, p2);  q2 = WMMA(a2, bf1, q2);
                p3 = WMMA(a3, bf0, p3);  q3 = WMMA(a3, bf1, q3);
            }
        }
        {
            const size_t pix = pixA;
            const float* cq0 = ctl + pix * 64;
            float* xq0 = xbuf + pix * 64;
            EPILOGUE(p0, 0) EPILOGUE(p1, 1) EPILOGUE(p2, 2) EPILOGUE(p3, 3)
        }
        {
            const size_t pix = pixB;
            const float* cq0 = ctl + pix * 64;
            float* xq0 = xbuf + pix * 64;
            EPILOGUE(q0, 0) EPILOGUE(q1, 1) EPILOGUE(q2, 2) EPILOGUE(q3, 3)
        }
    }
}

// ---------------------------------------------------------------------------
// NHWC f32 -> NCHW f32 final transpose via LDS tile (conflict-free, padded).
// ---------------------------------------------------------------------------
__global__ void to_nchw(const float* __restrict__ xbuf, float* __restrict__ out) {
    __shared__ float tile[64][65];
    const int x0 = blockIdx.x * 64;
    const int y  = blockIdx.y;
    const int b  = blockIdx.z;
    const int t  = threadIdx.x;
    {
        int pixl = t >> 2, cseg = (t & 3) * 16;
        const float* src = xbuf + (((size_t)b * HH + y) * WW + x0 + pixl) * 64 + cseg;
#pragma unroll
        for (int i = 0; i < 16; i += 4) {
            v4f v = *(const v4f*)(src + i);
            tile[pixl][cseg + i + 0] = v[0];
            tile[pixl][cseg + i + 1] = v[1];
            tile[pixl][cseg + i + 2] = v[2];
            tile[pixl][cseg + i + 3] = v[3];
        }
    }
    __syncthreads();
    {
        int c = t >> 2, xseg = (t & 3) * 16;
        float* dst = out + (((size_t)(b * 64 + c)) * HH + y) * WW + x0 + xseg;
#pragma unroll
        for (int i = 0; i < 16; i += 4) {
            v4f v;
            v[0] = tile[xseg + i + 0][c];
            v[1] = tile[xseg + i + 1][c];
            v[2] = tile[xseg + i + 2][c];
            v[3] = tile[xseg + i + 3][c];
            *(v4f*)(dst + i) = v;
        }
    }
}

// ---------------------------------------------------------------------------
extern "C" void kernel_launch(void* const* d_in, const int* in_sizes, int n_in,
                              void* d_out, int out_size, void* d_ws, size_t ws_size,
                              hipStream_t stream) {
    const float* u      = (const float*)d_in[0];
    const float* WB     = (const float*)d_in[1];
    const float* WA     = (const float*)d_in[2];
    const float* bias   = (const float*)d_in[3];
    const float* alogit = (const float*)d_in[4];
    float* out = (float*)d_out;

    char* ws = (char*)d_ws;
    const size_t fsz = NPIX * 64 * sizeof(float);      // 256 MB
    const size_t hsz = NPIX * 64 * sizeof(_Float16);   // 128 MB
    float*    ctl  = (float*)(ws);
    float*    xbuf = (float*)(ws + fsz);
    _Float16* ta   = (_Float16*)(ws + 2 * fsz);
    _Float16* tb   = (_Float16*)(ws + 2 * fsz + hsz);
    _Float16* wpk  = (_Float16*)(ws + 2 * fsz + 2 * hsz);

    prep_weights<<<9, 256, 0, stream>>>(WA, wpk);
    control_init<<<(unsigned)((NPIX * 4) / 256), 256, 0, stream>>>(u, WB, ctl, xbuf, ta);

    dim3 g(WW / TILE_W, HH / TILE_H, BB);
    const size_t lds = (size_t)6 * 130 * 64 * sizeof(_Float16);
    for (int it = 0; it < 16; ++it) {
        const _Float16* tin = (it & 1) ? tb : ta;
        _Float16*      tout = (it & 1) ? ta : tb;
        cenn_step<<<g, 256, lds, stream>>>(tin, tout, xbuf, ctl, bias, alogit, wpk);
    }

    to_nchw<<<dim3(WW / 64, HH, BB), 256, 0, stream>>>(xbuf, out);
}
